// GNN_50379966382774
// MI455X (gfx1250) — matrix operand order repaired and checked by reading, not compiled, not run on hardware
//
#include <hip/hip_runtime.h>

typedef __attribute__((ext_vector_type(16))) _Float16 v16h;
typedef __attribute__((ext_vector_type(8)))  float    v8f;

#define N_NODES  10000
#define N_EDGES  320000
#define NUM_FEAT 42
#define EDGE_FEAT 4
#define EMB 256
#define HID 512
#define NLAYER 3

// -------------------------------------------------------------------------
// One-time weight prep: f32 -> f16, swizzled into WMMA B-matrix tile layout.
// B tile (32x16 f16): lane = (N&15) | (Kin&16), element j = Kin&15.
// Tiles stored as (((nt*numKT + kt)*32 + lane)*16 + j) so a wave reads one
// contiguous v16h (32B) per lane.
// -------------------------------------------------------------------------
__global__ void prep_weights(const float* __restrict__ W1,
                             const float* __restrict__ W2,
                             _Float16* __restrict__ w1h,
                             _Float16* __restrict__ w2h) {
  int idx = blockIdx.x * blockDim.x + threadIdx.x;
  const int n1 = NLAYER * EMB * HID;
  if (idx < n1) {
    int l   = idx / (EMB * HID);
    int rem = idx - l * (EMB * HID);
    int K = rem / HID, N = rem - K * HID;
    int kt = K >> 5, kin = K & 31, nt = N >> 4;
    int lane = (N & 15) | (kin & 16);
    int j = kin & 15;
    w1h[l * (EMB * HID) + (((nt * 8 + kt) * 32 + lane) << 4) + j] = (_Float16)W1[idx];
  } else {
    int idx2 = idx - n1;
    if (idx2 < NLAYER * HID * EMB) {
      int l   = idx2 / (HID * EMB);
      int rem = idx2 - l * (HID * EMB);
      int K = rem / EMB, N = rem - K * EMB;
      int kt = K >> 5, kin = K & 31, nt = N >> 4;
      int lane = (N & 15) | (kin & 16);
      int j = kin & 15;
      w2h[l * (HID * EMB) + (((nt * 16 + kt) * 32 + lane) << 4) + j] = (_Float16)W2[idx2];
    }
  }
}

// -------------------------------------------------------------------------
// h = x @ W_feat + b_feat   (K=42, trivially small -> scalar FMA kernel)
// -------------------------------------------------------------------------
__global__ void feat_embed(const float* __restrict__ x,
                           const float* __restrict__ Wf,
                           const float* __restrict__ bf,
                           float* __restrict__ h) {
  __shared__ float xs[NUM_FEAT];
  int n = blockIdx.x;
  int c = threadIdx.x;
  if (c < NUM_FEAT) xs[c] = x[n * NUM_FEAT + c];
  __syncthreads();
  float acc = bf[c];
  for (int k = 0; k < NUM_FEAT; ++k) acc += xs[k] * Wf[k * EMB + c];
  h[n * EMB + c] = acc;
}

// -------------------------------------------------------------------------
// z = (1+eps[l]) * h ; also zero the BN stats accumulators
// -------------------------------------------------------------------------
__global__ void layer_init(const float* __restrict__ h,
                           const float* __restrict__ eps, int l,
                           float* __restrict__ z,
                           float* __restrict__ stats) {
  int idx = blockIdx.x * blockDim.x + threadIdx.x;
  float sc = 1.0f + eps[l];
  float4 v = ((const float4*)h)[idx];
  v.x *= sc; v.y *= sc; v.z *= sc; v.w *= sc;
  ((float4*)z)[idx] = v;
  if (blockIdx.x == 0 && threadIdx.x < 2 * EMB) stats[threadIdx.x] = 0.0f;
}

// -------------------------------------------------------------------------
// Per edge: msg = h[src] + (edge_attr @ We[l] + be[l]); atomic scatter to z[dst].
// One wave per edge, lane handles 8 channels. h/z live in L2 (10MB each).
// -------------------------------------------------------------------------
__global__ void edge_msg(const int* __restrict__ ei,
                         const float* __restrict__ ea,
                         const float* __restrict__ We,
                         const float* __restrict__ be,
                         const float* __restrict__ h,
                         float* __restrict__ z, int l) {
  int e = blockIdx.x * 8 + (threadIdx.x >> 5);
  if (e >= N_EDGES) return;
  int lane = threadIdx.x & 31;
  int src = ei[e];
  int dst = ei[N_EDGES + e];
  float a0 = ea[e * 4 + 0], a1 = ea[e * 4 + 1];
  float a2 = ea[e * 4 + 2], a3 = ea[e * 4 + 3];
  const float* Wl = We + l * 4 * EMB;
  const float* bl = be + l * EMB;
  int c0 = lane * 8;
#pragma unroll
  for (int half = 0; half < 2; ++half) {
    int c = c0 + half * 4;
    float4 w0 = *(const float4*)(Wl + 0 * EMB + c);
    float4 w1 = *(const float4*)(Wl + 1 * EMB + c);
    float4 w2 = *(const float4*)(Wl + 2 * EMB + c);
    float4 w3 = *(const float4*)(Wl + 3 * EMB + c);
    float4 bb = *(const float4*)(bl + c);
    float4 hv = *(const float4*)(h + (size_t)src * EMB + c);
    float m0 = hv.x + bb.x + a0 * w0.x + a1 * w1.x + a2 * w2.x + a3 * w3.x;
    float m1 = hv.y + bb.y + a0 * w0.y + a1 * w1.y + a2 * w2.y + a3 * w3.y;
    float m2 = hv.z + bb.z + a0 * w0.z + a1 * w1.z + a2 * w2.z + a3 * w3.z;
    float m3 = hv.w + bb.w + a0 * w0.w + a1 * w1.w + a2 * w2.w + a3 * w3.w;
    float* zr = z + (size_t)dst * EMB + c;
    atomicAdd(zr + 0, m0);
    atomicAdd(zr + 1, m1);
    atomicAdd(zr + 2, m2);
    atomicAdd(zr + 3, m3);
  }
}

// -------------------------------------------------------------------------
// Fused GIN MLP per 16-node tile, via v_wmma_f32_16x16x32_f16.
//   z1 = relu(z @ W1 + b1)  [16x512]   (GEMM1, K=256, 8 waves x 4 N-tiles)
//   zo = z1 @ W2 + b2       [16x256]   (GEMM2, K=512, 8 waves x 2 N-tiles)
// In-place on z; also accumulates per-channel sum / sum^2 for BatchNorm.
// A-matrix 16x32 f16 layout used here: lane<16 holds K j (j=0..15),
// lane>=16 holds K 16+j; M = lane&15.
// -------------------------------------------------------------------------
__global__ void __launch_bounds__(256)
gin_mlp_wmma(const float* __restrict__ zin,
             const _Float16* __restrict__ w1h,
             const _Float16* __restrict__ w2h,
             const float* __restrict__ b1,
             const float* __restrict__ b2,
             float* __restrict__ zout,
             float* __restrict__ stats, int l) {
  __shared__ __align__(32) _Float16 aA[16 * EMB];   // 16x256 A tiles (8 k-tiles)
  __shared__ __align__(32) _Float16 aA2[16 * HID];  // 16x512 A tiles (16 k-tiles)
  __shared__ float lsum[EMB];
  __shared__ float lsum2[EMB];

  int tid = threadIdx.x;
  int m0 = blockIdx.x * 16;
  if (tid < EMB) { lsum[tid] = 0.0f; lsum2[tid] = 0.0f; }

  // Build A1: each thread owns one (k-tile, lane) slot = 16 halves.
  {
    int kt = tid >> 5, lane = tid & 31;
    int row = lane & 15;
    const float* zr = zin + (size_t)(m0 + row) * EMB;
#pragma unroll
    for (int j = 0; j < 16; ++j) {
      int K = (kt << 5) + (lane & 16) + j;
      aA[(tid << 4) + j] = (_Float16)zr[K];
    }
  }
  __syncthreads();

  int lane = tid & 31;
  int wv = tid >> 5;
  const _Float16* w1l = w1h + l * (EMB * HID);
  const _Float16* w2l = w2h + l * (HID * EMB);
  const float* b1l = b1 + l * HID;
  const float* b2l = b2 + l * EMB;
  int rowbase = (lane >> 4) << 3;  // C/D layout: VGPR r -> row r (+8 for lanes>=16)
  int ncol = lane & 15;

  // ---- GEMM1 + bias + ReLU, written back into LDS in A-layout for GEMM2 ----
  for (int nt = wv * 4; nt < wv * 4 + 4; ++nt) {
    v8f acc = {};
#pragma unroll
    for (int kt = 0; kt < 8; ++kt) {
      v16h a = *(const v16h*)(aA + (kt << 9) + (lane << 4));
      v16h b = *(const v16h*)(w1l + (((nt * 8 + kt) * 32 + lane) << 4));
      acc = __builtin_amdgcn_wmma_f32_16x16x32_f16(false, a, false, b,
                                                   (short)0, acc, false, false);
    }
    int col = (nt << 4) + ncol;            // global column in [0,512)
    float bias = b1l[col];
    int kt2 = col >> 5, kin = col & 31;
    int lane2hi = kin & 16;
    int jj = kin & 15;
#pragma unroll
    for (int r = 0; r < 8; ++r) {
      float v = acc[r] + bias;
      v = v > 0.0f ? v : 0.0f;
      int row = rowbase + r;
      aA2[(kt2 << 9) + ((row | lane2hi) << 4) + jj] = (_Float16)v;
    }
  }
  __syncthreads();

  // ---- GEMM2 + bias, store + BN partial stats ----
  for (int nt = wv * 2; nt < wv * 2 + 2; ++nt) {
    v8f acc = {};
#pragma unroll
    for (int kt = 0; kt < 16; ++kt) {
      v16h a = *(const v16h*)(aA2 + (kt << 9) + (lane << 4));
      v16h b = *(const v16h*)(w2l + (((nt * 16 + kt) * 32 + lane) << 4));
      acc = __builtin_amdgcn_wmma_f32_16x16x32_f16(false, a, false, b,
                                                   (short)0, acc, false, false);
    }
    int col = (nt << 4) + ncol;            // global column in [0,256)
    float bias = b2l[col];
    float s = 0.0f, s2 = 0.0f;
#pragma unroll
    for (int r = 0; r < 8; ++r) {
      float v = acc[r] + bias;
      zout[(size_t)(m0 + rowbase + r) * EMB + col] = v;
      s += v; s2 += v * v;
    }
    atomicAdd(&lsum[col], s);
    atomicAdd(&lsum2[col], s2);
  }
  __syncthreads();
  if (tid < EMB) {
    atomicAdd(&stats[tid], lsum[tid]);
    atomicAdd(&stats[EMB + tid], lsum2[tid]);
  }
}

// -------------------------------------------------------------------------
// Turn BN stats into per-channel affine: out = z*a + b, then ReLU.
// -------------------------------------------------------------------------
__global__ void bn_prepare(const float* __restrict__ stats,
                           const float* __restrict__ gamma,
                           const float* __restrict__ beta,
                           float* __restrict__ ab, int l) {
  int c = threadIdx.x;
  const float inv_n = 1.0f / (float)N_NODES;
  float mean = stats[c] * inv_n;
  float ex2  = stats[EMB + c] * inv_n;
  float var = ex2 - mean * mean;
  float a = gamma[l * EMB + c] * rsqrtf(var + 1e-5f);
  float b = beta[l * EMB + c] - mean * a;
  ab[c] = a;
  ab[EMB + c] = b;
}

__global__ void bn_apply(const float* __restrict__ z,
                         const float* __restrict__ ab,
                         float* __restrict__ hout) {
  int idx = blockIdx.x * blockDim.x + threadIdx.x;
  int c0 = (idx << 2) & (EMB - 1);
  float4 v = ((const float4*)z)[idx];
  float4 A = *(const float4*)(ab + c0);
  float4 B = *(const float4*)(ab + EMB + c0);
  float4 o;
  o.x = fmaxf(v.x * A.x + B.x, 0.0f);
  o.y = fmaxf(v.y * A.y + B.y, 0.0f);
  o.z = fmaxf(v.z * A.z + B.z, 0.0f);
  o.w = fmaxf(v.w * A.w + B.w, 0.0f);
  ((float4*)hout)[idx] = o;
}

// -------------------------------------------------------------------------
extern "C" void kernel_launch(void* const* d_in, const int* in_sizes, int n_in,
                              void* d_out, int out_size, void* d_ws, size_t ws_size,
                              hipStream_t stream) {
  (void)in_sizes; (void)n_in; (void)out_size; (void)ws_size;
  const float* x     = (const float*)d_in[0];
  const int*   ei    = (const int*)d_in[1];
  const float* ea    = (const float*)d_in[2];
  const float* Wf    = (const float*)d_in[3];
  const float* bf    = (const float*)d_in[4];
  const float* eps   = (const float*)d_in[5];
  const float* We    = (const float*)d_in[6];
  const float* be    = (const float*)d_in[7];
  const float* W1    = (const float*)d_in[8];
  const float* b1    = (const float*)d_in[9];
  const float* W2    = (const float*)d_in[10];
  const float* b2    = (const float*)d_in[11];
  const float* gamma = (const float*)d_in[12];
  const float* beta  = (const float*)d_in[13];
  float* out = (float*)d_out;

  // Workspace layout (all 256B-aligned offsets), ~22.1 MB total:
  char* ws = (char*)d_ws;
  float*    h     = (float*)(ws);                      // 10,240,000 B
  float*    z     = (float*)(ws + 10240000);           // 10,240,000 B
  _Float16* w1h   = (_Float16*)(ws + 20480000);        //    786,432 B
  _Float16* w2h   = (_Float16*)(ws + 21266432);        //    786,432 B
  float*    stats = (float*)(ws + 22052864);           // 2*256 sums + 2*256 affine
  float*    ab    = stats + 2 * EMB;

  prep_weights<<<3072, 256, 0, stream>>>(W1, W2, w1h, w2h);
  feat_embed<<<N_NODES, 256, 0, stream>>>(x, Wf, bf, h);

  for (int l = 0; l < NLAYER; ++l) {
    layer_init<<<(N_NODES * EMB) / (4 * 256), 256, 0, stream>>>(h, eps, l, z, stats);
    edge_msg<<<N_EDGES / 8, 256, 0, stream>>>(ei, ea, We, be, h, z, l);
    gin_mlp_wmma<<<N_NODES / 16, 256, 0, stream>>>(z, w1h, w2h, b1, b2, z, stats, l);
    bn_prepare<<<1, 256, 0, stream>>>(stats, gamma, beta, ab, l);
    float* dst = (l == NLAYER - 1) ? out : h;
    bn_apply<<<(N_NODES * EMB) / (4 * 256), 256, 0, stream>>>(z, ab, dst);
  }
}
